// WordConv_6133213299286
// MI455X (gfx1250) — compile-verified, matrix operand order checked by reading, not run on hardware
//
#include <hip/hip_runtime.h>

typedef __attribute__((ext_vector_type(16))) _Float16 v16h;
typedef __attribute__((ext_vector_type(8)))  _Float16 v8h;
typedef __attribute__((ext_vector_type(8)))  float    v8f;
typedef __attribute__((ext_vector_type(4)))  float    v4f;

#define VOCAB 2048
#define NG    48
#define KD    128
#define NB    1024

// ---------------------------------------------------------------------------
// Kernel 1a: word_idx[b] = argmax of one-hot row = dot(row, iota). Exact in f32.
// One wave (32 lanes) per b.
// ---------------------------------------------------------------------------
__global__ __launch_bounds__(128) void k_widx(const float* __restrict__ inp,
                                              int* __restrict__ widx) {
    const int wave = threadIdx.x >> 5;
    const int lane = threadIdx.x & 31;
    const int b = blockIdx.x * 4 + wave;
    const float* row = inp + (size_t)b * VOCAB;
    float s = 0.0f;
#pragma unroll 4
    for (int i = 0; i < VOCAB / 32; ++i) {
        const int v = lane + i * 32;
        s += row[v] * (float)v;
    }
    for (int off = 16; off > 0; off >>= 1) s += __shfl_xor(s, off, 32);
    if (lane == 0) widx[b] = (int)(s + 0.5f);
}

// ---------------------------------------------------------------------------
// Kernel 1b: embT[k][v] = (f16) emb[v][k]  (512 KB, lives in workspace / L2).
// Coalesced writes; reads are L2-hot (emb is 1 MB).
// ---------------------------------------------------------------------------
__global__ __launch_bounds__(256) void k_embT(const float* __restrict__ emb,
                                              _Float16* __restrict__ embT) {
    const int idx = blockIdx.x * 256 + threadIdx.x;   // idx = k*VOCAB + v
    const int k = idx / VOCAB;
    const int v = idx - k * VOCAB;
    embT[idx] = (_Float16)emb[(size_t)v * KD + k];
}

__device__ inline v16h cat8(v8h lo, v8h hi) {
    v16h r;
#pragma unroll
    for (int i = 0; i < 8; ++i) { r[i] = lo[i]; r[i + 8] = hi[i]; }
    return r;
}

// ---------------------------------------------------------------------------
// Kernel 2: out[b,g,:] = perms[g, widx[b], :] @ emb   via v_wmma_f32_16x16x32_f16
// Block = 256 threads = 8 waves; block computes a 64(row) x 128(K) f32 tile for one g.
// Wave (strip = w>>1, half = w&1) owns rows strip*16..+15, cols half*64..+63
// as 4 accumulators of 16x16.
//
// Fragment layouts per CDNA5 ISA 7.12.2:
//   A (16x32 f16): lane m(0-15): K runs {0..7}(v0-3) {16..23}(v4-7); lanes 16-31: {8..15},{24..31}
//   B (32x16 f16): lane n(0-15) holds column n with the same K-run pattern
//   C (16x16 f32): VGPR j: lanes 0-15 -> M=j, lanes 16-31 -> M=j+8, N=lane&15
// ---------------------------------------------------------------------------
__global__ __launch_bounds__(256) void k_gemm(const float* __restrict__ perms,
                                              const _Float16* __restrict__ embT,
                                              const int* __restrict__ widx,
                                              float* __restrict__ out) {
    const int g    = blockIdx.y;
    const int b0   = blockIdx.x * 64;
    const int wave = threadIdx.x >> 5;
    const int lane = threadIdx.x & 31;
    const int strip = wave >> 1;     // 0..3 : 16-row strip
    const int half  = wave & 1;      // 0..1 : 64-col half
    const int n  = lane & 15;
    const int hi = lane >> 4;
    const int k0 = half * 64;

    // Gathered A row for this lane's M index (lanes m and m+16 share row m).
    const int ridx = widx[b0 + strip * 16 + n];
    const float* rowp = perms + ((size_t)g * VOCAB + (size_t)ridx) * VOCAB;

    // B column pointers (f16, contiguous in K) — L2-resident.
    const _Float16* colp0 = embT + (size_t)(k0 +  0 + n) * VOCAB;
    const _Float16* colp1 = embT + (size_t)(k0 + 16 + n) * VOCAB;
    const _Float16* colp2 = embT + (size_t)(k0 + 32 + n) * VOCAB;
    const _Float16* colp3 = embT + (size_t)(k0 + 48 + n) * VOCAB;

    v8f acc0 = {}, acc1 = {}, acc2 = {}, acc3 = {};

    for (int kc = 0; kc < VOCAB; kc += 32) {
        const int r1 = kc + hi * 8;        // first 8-wide K run for this lane
        const int r2 = kc + 16 + hi * 8;   // second 8-wide K run

        // --- A fragment: 16 f32 -> 16 f16 (exact: values are 0.0 / 1.0) ---
        v4f a0 = *(const v4f*)(rowp + r1);
        v4f a1 = *(const v4f*)(rowp + r1 + 4);
        v4f a2 = *(const v4f*)(rowp + r2);
        v4f a3 = *(const v4f*)(rowp + r2 + 4);
        v16h afrag;
#pragma unroll
        for (int i = 0; i < 4; ++i) {
            afrag[i]      = (_Float16)a0[i];
            afrag[i + 4]  = (_Float16)a1[i];
            afrag[i + 8]  = (_Float16)a2[i];
            afrag[i + 12] = (_Float16)a3[i];
        }

        // --- B fragments: straight 16B f16 loads from embT ---
        v16h bf0 = cat8(*(const v8h*)(colp0 + r1), *(const v8h*)(colp0 + r2));
        v16h bf1 = cat8(*(const v8h*)(colp1 + r1), *(const v8h*)(colp1 + r2));
        v16h bf2 = cat8(*(const v8h*)(colp2 + r1), *(const v8h*)(colp2 + r2));
        v16h bf3 = cat8(*(const v8h*)(colp3 + r1), *(const v8h*)(colp3 + r2));

        acc0 = __builtin_amdgcn_wmma_f32_16x16x32_f16(false, afrag, false, bf0,
                                                      (short)0, acc0, false, false);
        acc1 = __builtin_amdgcn_wmma_f32_16x16x32_f16(false, afrag, false, bf1,
                                                      (short)0, acc1, false, false);
        acc2 = __builtin_amdgcn_wmma_f32_16x16x32_f16(false, afrag, false, bf2,
                                                      (short)0, acc2, false, false);
        acc3 = __builtin_amdgcn_wmma_f32_16x16x32_f16(false, afrag, false, bf3,
                                                      (short)0, acc3, false, false);
    }

    // --- Store C tiles: out[b, g, k] with b = b0 + strip*16 + hi*8 + j ---
#pragma unroll
    for (int j = 0; j < 8; ++j) {
        const int m = strip * 16 + hi * 8 + j;
        float* o = out + (size_t)(b0 + m) * (NG * KD) + (size_t)g * KD + k0 + n;
        o[0]  = acc0[j];
        o[16] = acc1[j];
        o[32] = acc2[j];
        o[48] = acc3[j];
    }
}

// ---------------------------------------------------------------------------
extern "C" void kernel_launch(void* const* d_in, const int* in_sizes, int n_in,
                              void* d_out, int out_size, void* d_ws, size_t ws_size,
                              hipStream_t stream) {
    const float* inp   = (const float*)d_in[0];   // [B, V] one-hot
    const float* perms = (const float*)d_in[1];   // [G, V, V]
    const float* emb   = (const float*)d_in[2];   // [V, K]
    float* out = (float*)d_out;                   // [B, G, K] f32

    int* widx = (int*)d_ws;                                   // B ints
    _Float16* embT = (_Float16*)((char*)d_ws + 4096);         // K*V f16 = 512 KB

    k_widx<<<NB / 4, 128, 0, stream>>>(inp, widx);
    k_embT<<<(KD * VOCAB) / 256, 256, 0, stream>>>(emb, embT);

    dim3 grid(NB / 64, NG);   // b-tiles fastest -> same-g blocks adjacent (L2 reuse)
    k_gemm<<<grid, 256, 0, stream>>>(perms, embT, widx, out);
}
